// MoE_74689481277447
// MI455X (gfx1250) — compile-verified, matrix operand order checked by reading, not run on hardware
//
#include <hip/hip_runtime.h>
#include <hip/hip_bf16.h>
#include <math.h>

// ---------------------------------------------------------------------------
// MoE (top-2 of 8 experts), D=1024, F=4096, N=8192 tokens.
// bf16 WMMA GEMMs on gathered per-expert segments, double-buffered LDS.
// Weights pre-transposed to (E, NC, K) bf16 so GEMM staging needs no VALU.
// ---------------------------------------------------------------------------

#define NT      8192          // B*T tokens
#define DM      1024          // d_model
#define FF      4096          // d_ff
#define NE      8             // experts
#define TOPK    2
#define PADMAX  (NT*TOPK + NE*128)   // 17408 padded routed rows

typedef __attribute__((ext_vector_type(16))) __bf16 v16bf;
typedef __attribute__((ext_vector_type(8)))  float  v8f;

__device__ __forceinline__ unsigned short f2bf(float f) {
    union { float f; unsigned u; } v; v.f = f;
    unsigned r = v.u + 0x7FFFu + ((v.u >> 16) & 1u);   // round-nearest-even
    return (unsigned short)(r >> 16);
}

__device__ __forceinline__ float fast_sigmoid(float z) {
    return 1.f / (1.f + __expf(-z));
}

// tanh-GELU == v*sigmoid(2*0.79788456*(v+0.044715*v^3))
__device__ __forceinline__ float fast_gelu(float v) {
    float u = 1.5957691216057308f * (v + 0.044715f * v * v * v);
    return v * fast_sigmoid(u);
}

// ---------------------------------------------------------------------------
// 1) init: zero counts, rowmap = -1
// ---------------------------------------------------------------------------
__global__ void moe_init_kernel(int* __restrict__ counts, int* __restrict__ rowmap) {
    int i = blockIdx.x * blockDim.x + threadIdx.x;
    if (i < PADMAX) rowmap[i] = -1;
    if (i < NE) counts[i] = 0;
}

// ---------------------------------------------------------------------------
// 2) gate: one wave per token -> 8 dots over D=1024, sigmoid, top-2
// ---------------------------------------------------------------------------
__global__ __launch_bounds__(256) void moe_gate_kernel(
    const float* __restrict__ x, const float* __restrict__ Wg,
    const float* __restrict__ bg, float* __restrict__ gout,
    int* __restrict__ texp, float* __restrict__ wslot, int* __restrict__ counts)
{
    const int lane = threadIdx.x & 31;
    const int w    = threadIdx.x >> 5;
    const int t    = blockIdx.x * 8 + w;
    if (t >= NT) return;

    float acc[NE];
#pragma unroll
    for (int e = 0; e < NE; ++e) acc[e] = 0.f;

    const float* xr = x + (size_t)t * DM;
    for (int d = lane; d < DM; d += 32) {
        float xv = xr[d];
        const float* wr = Wg + (size_t)d * NE;
#pragma unroll
        for (int e = 0; e < NE; ++e) acc[e] += xv * wr[e];
    }
#pragma unroll
    for (int off = 16; off > 0; off >>= 1) {
#pragma unroll
        for (int e = 0; e < NE; ++e) acc[e] += __shfl_xor(acc[e], off, 32);
    }
    if (lane == 0) {
        float s[NE];
#pragma unroll
        for (int e = 0; e < NE; ++e) {
            s[e] = fast_sigmoid(acc[e] + bg[e]);
            gout[(size_t)t * NE + e] = s[e];
        }
        int i0 = 0;
#pragma unroll
        for (int e = 1; e < NE; ++e) if (s[e] > s[i0]) i0 = e;
        int i1 = (i0 == 0) ? 1 : 0;
#pragma unroll
        for (int e = 0; e < NE; ++e) if (e != i0 && s[e] > s[i1]) i1 = e;
        texp[2 * t]     = i0;  wslot[2 * t]     = s[i0];
        texp[2 * t + 1] = i1;  wslot[2 * t + 1] = s[i1];
        atomicAdd(&counts[i0], 1);
        atomicAdd(&counts[i1], 1);
    }
}

// ---------------------------------------------------------------------------
// 3) scan: 128-aligned exclusive prefix over 8 expert counts
// ---------------------------------------------------------------------------
__global__ void moe_scan_kernel(const int* __restrict__ counts,
                                int* __restrict__ offs, int* __restrict__ cursors) {
    if (threadIdx.x == 0 && blockIdx.x == 0) {
        int acc = 0;
        for (int e = 0; e < NE; ++e) {
            offs[e] = acc;
            cursors[e] = acc;
            acc += (counts[e] + 127) & ~127;
        }
        offs[NE] = acc;
    }
}

// ---------------------------------------------------------------------------
// 4) scatter: assign routed rows inside expert segments
// ---------------------------------------------------------------------------
__global__ void moe_scatter_kernel(const int* __restrict__ texp,
                                   int* __restrict__ cursors,
                                   int* __restrict__ rowmap,
                                   int* __restrict__ tok2row) {
    int idx = blockIdx.x * blockDim.x + threadIdx.x;
    if (idx >= NT * TOPK) return;
    int e = texp[idx];
    int r = atomicAdd(&cursors[e], 1);
    rowmap[r] = idx >> 1;      // token id
    tok2row[idx] = r;
}

// ---------------------------------------------------------------------------
// 5a) gather x rows into padded routed layout, fp32 -> bf16
// ---------------------------------------------------------------------------
__global__ __launch_bounds__(256) void moe_gather_kernel(
    const float* __restrict__ x, const int* __restrict__ rowmap,
    unsigned short* __restrict__ xg)
{
    int idx = blockIdx.x * blockDim.x + threadIdx.x;       // PADMAX * 128 threads
    int row = idx >> 7;
    int c0  = (idx & 127) << 3;
    if (row >= PADMAX) return;
    int tok = rowmap[row];
    unsigned short* dst = xg + (size_t)row * DM + c0;
    if (tok < 0) {
        uint4 z = {0, 0, 0, 0};
        *(uint4*)dst = z;
        return;
    }
    const float4* src = (const float4*)(x + (size_t)tok * DM + c0);
    float4 a = src[0], b = src[1];
    unsigned p[4];
    p[0] = (unsigned)f2bf(a.x) | ((unsigned)f2bf(a.y) << 16);
    p[1] = (unsigned)f2bf(a.z) | ((unsigned)f2bf(a.w) << 16);
    p[2] = (unsigned)f2bf(b.x) | ((unsigned)f2bf(b.y) << 16);
    p[3] = (unsigned)f2bf(b.z) | ((unsigned)f2bf(b.w) << 16);
    *(uint4*)dst = make_uint4(p[0], p[1], p[2], p[3]);
}

// ---------------------------------------------------------------------------
// 5b) weight convert + transpose: (E,K,NC) fp32 -> (E,NC,K) bf16.
//     64x64 tiles through LDS; coalesced global read and write.
// ---------------------------------------------------------------------------
__global__ __launch_bounds__(256) void moe_cvtT_kernel(
    const float* __restrict__ src, unsigned short* __restrict__ dst,
    int K, int NC)
{
    __shared__ unsigned short T[64][68];    // pad 4 -> row stride 136 B (8B aligned)
    const int e  = blockIdx.z;
    const int n0 = blockIdx.x * 64;
    const int k0 = blockIdx.y * 64;
    const int tid = threadIdx.x;

#pragma unroll
    for (int it = 0; it < 4; ++it) {
        int k_l = (tid >> 4) + 16 * it;
        int n_l = (tid & 15) * 4;
        float4 v = *(const float4*)&src[((size_t)e * K + k0 + k_l) * NC + n0 + n_l];
        T[n_l + 0][k_l] = f2bf(v.x);
        T[n_l + 1][k_l] = f2bf(v.y);
        T[n_l + 2][k_l] = f2bf(v.z);
        T[n_l + 3][k_l] = f2bf(v.w);
    }
    __syncthreads();
#pragma unroll
    for (int it = 0; it < 4; ++it) {
        int n_l = (tid >> 4) + 16 * it;
        int k_l = (tid & 15) * 4;
        uint2 pk = *(const uint2*)&T[n_l][k_l];
        *(uint2*)&dst[((size_t)e * NC + n0 + n_l) * K + k0 + k_l] = pk;
    }
}

// ---------------------------------------------------------------------------
// 6/7) bf16 WMMA GEMM over expert segments, double-buffered LDS.
//   A: padded rows x K (bf16 row-major), B: per-expert NC x K (bf16, K-major).
//   Workgroup tile 128(M) x 256(N); 8 waves, each 64x64 (4x4 WMMA 16x16x32).
//   One barrier per K-step; next tile loaded to regs during compute.
// ---------------------------------------------------------------------------
template<int K, int NC, bool GELU>
__global__ __launch_bounds__(256) void moe_gemm_kernel(
    const unsigned short* __restrict__ A,
    const unsigned short* __restrict__ Bw,
    const float* __restrict__ bias,
    unsigned short* __restrict__ Hout,
    float* __restrict__ Fout,
    const int* __restrict__ offs)
{
    __shared__ alignas(64) unsigned short As[2][128 * 32];   // 2 x  8 KB
    __shared__ alignas(64) unsigned short Bs[2][256 * 32];   // 2 x 16 KB, [n][k]

    const int m0 = blockIdx.x * 128;
    if (m0 >= offs[NE]) return;
    int e = 0;
    while (e < NE - 1 && m0 >= offs[e + 1]) ++e;

    const int n0   = blockIdx.y * 256;
    const unsigned short* Bbase = Bw + (size_t)e * NC * K;

    const int tid  = threadIdx.x;
    const int lane = tid & 31;
    const int w    = tid >> 5;
    const int wm   = w & 1;        // 0..1 -> 64-row half
    const int wn   = w >> 1;       // 0..3 -> 64-col quarter

    v8f acc[4][4] = {};

    // A-load mapping: 128 rows x 32 k, 16 bf16 per thread
    const int ar = tid >> 1, ak = (tid & 1) * 16;
    const unsigned short* Aln = A + (size_t)(m0 + ar) * K + ak;
    // B-load mapping: 1 n-column x 16 k per thread (K-major weights)
    const unsigned short* Bln = Bbase + (size_t)(n0 + tid) * K;

    uint4 sa0, sa1, sb0, sb1;      // staged regs

    auto load_tiles = [&](int kt) {
        const uint4* asrc = (const uint4*)(Aln + kt);
        sa0 = asrc[0];
        sa1 = asrc[1];
        const uint4* bsrc = (const uint4*)(Bln + kt);
        sb0 = bsrc[0];
        sb1 = bsrc[1];
    };
    auto store_tiles = [&](int p) {
        *(uint4*)(&As[p][ar * 32 + ak])      = sa0;
        *(uint4*)(&As[p][ar * 32 + ak + 8])  = sa1;
        *(uint4*)(&Bs[p][tid * 32])          = sb0;
        *(uint4*)(&Bs[p][tid * 32 + 8])      = sb1;
    };

    load_tiles(0);
    store_tiles(0);

    int p = 0;
    for (int kt = 0; kt < K; kt += 32) {
        const bool more = (kt + 32) < K;
        if (more) load_tiles(kt + 32);
        if ((kt + 64) < K) {
            // warm L2 two K-steps ahead (global_prefetch_b8)
            __builtin_prefetch(Aln + kt + 64, 0, 1);
            __builtin_prefetch(Bln + kt + 64, 0, 1);
        }
        __syncthreads();   // tile(kt) visible; everyone done computing tile(kt-32)

        // --- preload all 8 fragments, then 16 WMMAs (staggered ds waits) ---
        v16bf af[4], bf[4];
#pragma unroll
        for (int i = 0; i < 4; ++i) {
            int row = 64 * wm + 16 * i + (lane & 15);
            af[i] = *(const v16bf*)&As[p][row * 32 + (lane >> 4) * 16];
        }
#pragma unroll
        for (int j = 0; j < 4; ++j) {
            int col = 64 * wn + 16 * j + (lane & 15);
            bf[j] = *(const v16bf*)&Bs[p][col * 32 + (lane >> 4) * 16];
        }
#pragma unroll
        for (int j = 0; j < 4; ++j) {
#pragma unroll
            for (int i = 0; i < 4; ++i) {
                acc[i][j] = __builtin_amdgcn_wmma_f32_16x16x32_bf16(
                    false, af[i], false, bf[j], (short)0, acc[i][j], false, false);
            }
        }

        if (more) store_tiles(p ^ 1);
        p ^= 1;
    }

    // --- epilogue: C layout = 8 VGPRs; lane<16: M=vgpr, lane>=16: M=vgpr+8 ---
#pragma unroll
    for (int j = 0; j < 4; ++j) {
        const int nb = n0 + 64 * wn + 16 * j + (lane & 15);
        const float bv = bias[(size_t)e * NC + nb];
#pragma unroll
        for (int i = 0; i < 4; ++i) {
            const int mb = m0 + 64 * wm + 16 * i + ((lane >> 4) << 3);
#pragma unroll
            for (int r = 0; r < 8; ++r) {
                float v = acc[i][j][r] + bv;
                const size_t o = (size_t)(mb + r) * NC + nb;
                if constexpr (GELU) {
                    Hout[o] = f2bf(fast_gelu(v));
                } else {
                    Fout[o] = v;
                }
            }
        }
    }
}

// ---------------------------------------------------------------------------
// 8) combine: y[t] = w0*yg[r0] + w1*yg[r1]
// ---------------------------------------------------------------------------
__global__ __launch_bounds__(256) void moe_combine_kernel(
    const float* __restrict__ yg, const int* __restrict__ tok2row,
    const float* __restrict__ wslot, float* __restrict__ y)
{
    int idx = blockIdx.x * blockDim.x + threadIdx.x;   // NT*256 threads
    int t = idx >> 8;
    int c = (idx & 255) << 2;
    if (t >= NT) return;
    int   r0 = tok2row[2 * t],     r1 = tok2row[2 * t + 1];
    float w0 = wslot[2 * t],       w1 = wslot[2 * t + 1];
    float4 a = *(const float4*)(yg + (size_t)r0 * DM + c);
    float4 b = *(const float4*)(yg + (size_t)r1 * DM + c);
    float4 o;
    o.x = w0 * a.x + w1 * b.x;
    o.y = w0 * a.y + w1 * b.y;
    o.z = w0 * a.z + w1 * b.z;
    o.w = w0 * a.w + w1 * b.w;
    *(float4*)(y + (size_t)t * DM + c) = o;
}

// ---------------------------------------------------------------------------
// launcher
// ---------------------------------------------------------------------------
extern "C" void kernel_launch(void* const* d_in, const int* in_sizes, int n_in,
                              void* d_out, int out_size, void* d_ws, size_t ws_size,
                              hipStream_t stream) {
    const float* x  = (const float*)d_in[0];
    const float* Wg = (const float*)d_in[1];
    const float* bg = (const float*)d_in[2];
    const float* W1 = (const float*)d_in[3];
    const float* b1 = (const float*)d_in[4];
    const float* W2 = (const float*)d_in[5];
    const float* b2 = (const float*)d_in[6];
    float* y    = (float*)d_out;
    float* gout = y + (size_t)NT * DM;

    const size_t WSZ = (size_t)NE * DM * FF;       // elements per weight tensor

    unsigned short* W1b = (unsigned short*)d_ws;           // (E, FF, DM) bf16
    unsigned short* W2b = W1b + WSZ;                       // (E, DM, FF) bf16
    unsigned short* xg  = W2b + WSZ;                       // PADMAX x DM
    unsigned short* h   = xg + (size_t)PADMAX * DM;        // PADMAX x FF
    float* yg           = (float*)(h + (size_t)PADMAX * FF);
    int* counts   = (int*)(yg + (size_t)PADMAX * DM);
    int* offs     = counts + NE;        // NE+1
    int* cursors  = offs + NE + 1;
    int* rowmap   = cursors + NE;       // PADMAX
    int* texp     = rowmap + PADMAX;    // NT*2
    int* tok2row  = texp + NT * TOPK;   // NT*2
    float* wslot  = (float*)(tok2row + NT * TOPK);

    moe_init_kernel<<<(PADMAX + 255) / 256, 256, 0, stream>>>(counts, rowmap);
    moe_gate_kernel<<<NT / 8, 256, 0, stream>>>(x, Wg, bg, gout, texp, wslot, counts);
    moe_scan_kernel<<<1, 32, 0, stream>>>(counts, offs, cursors);
    moe_scatter_kernel<<<(NT * TOPK + 255) / 256, 256, 0, stream>>>(texp, cursors, rowmap, tok2row);
    moe_gather_kernel<<<(PADMAX * 128 + 255) / 256, 256, 0, stream>>>(x, rowmap, xg);

    // W1: (E, DM, FF) -> (E, FF, DM);  W2: (E, FF, DM) -> (E, DM, FF)
    dim3 c1(FF / 64, DM / 64, NE);
    moe_cvtT_kernel<<<c1, 256, 0, stream>>>(W1, W1b, DM, FF);
    dim3 c2(DM / 64, FF / 64, NE);
    moe_cvtT_kernel<<<c2, 256, 0, stream>>>(W2, W2b, FF, DM);

    dim3 g1(PADMAX / 128, FF / 256);   // 136 x 16
    moe_gemm_kernel<DM, FF, true><<<g1, 256, 0, stream>>>(xg, W1b, b1, h, nullptr, offs);
    dim3 g2(PADMAX / 128, DM / 256);   // 136 x 4
    moe_gemm_kernel<FF, DM, false><<<g2, 256, 0, stream>>>(h, W2b, b2, nullptr, yg, offs);

    moe_combine_kernel<<<NT * 256 / 256, 256, 0, stream>>>(yg, tok2row, wslot, y);
}
